// FlagBagEncoder_53163105190342
// MI455X (gfx1250) — matrix operand order repaired and checked by reading, not compiled, not run on hardware
//
#include <hip/hip_runtime.h>
#include <hip/hip_bf16.h>

typedef __attribute__((ext_vector_type(16))) _Float16 v16h;
typedef __attribute__((ext_vector_type(8)))  float    v8f;

#define K_DIM 512
#define D_DIM 64
#define WAVES_PER_BLOCK 4
#define BLOCK_THREADS (WAVES_PER_BLOCK * 32)

// One wave computes a 16x64 output tile:
//   C[16x64] = mask_f16[16x512] @ emb_f16[512x64], fp32 accumulate.
// A 5th WMMA accumulator against an all-ones B gives per-row active counts
// (count[row] replicated across all 16 columns -> no cross-lane shuffles).
__global__ __launch_bounds__(BLOCK_THREADS)
void FlagBagEncoder_wmma_kernel(const float* __restrict__ flags,
                                const float* __restrict__ emb,
                                float* __restrict__ out,
                                int T) {
    // emb staged as f16, transposed: ldsEmbT[d][k], 64 KB (LDS is 320KB/WGP on CDNA5)
    __shared__ _Float16 ldsEmbT[D_DIM * K_DIM];

    const int tid = threadIdx.x;

    // ---- Stage emb (f32 [K,D] row-major) -> LDS f16 [D][K] (coalesced global reads) ----
    for (int i = tid; i < K_DIM * D_DIM; i += BLOCK_THREADS) {
        const int k = i >> 6;   // i / D_DIM
        const int d = i & 63;   // i % D_DIM
        ldsEmbT[d * K_DIM + k] = (_Float16)emb[i];
    }
    __syncthreads();

    const int lane = tid & 31;
    const int wave = tid >> 5;
    const int g    = lane >> 4;   // half-wave group (0: lanes 0-15, 1: lanes 16-31)
    const int n    = lane & 15;   // N index / A-row index for this lane

    const int numTiles = T >> 4;                      // 100000/16 = 6250
    const int tile     = blockIdx.x * WAVES_PER_BLOCK + wave;
    if (tile >= numTiles) return;                     // wave-uniform: EXEC stays all-1s for WMMA

    const int    row0 = tile * 16;
    const float* pRow = flags + (size_t)(row0 + n) * K_DIM;   // lane n loads A row n

    // All-ones B fragment (32x16 of f16 1.0): C += mask @ ones -> per-row counts in every column
    v16h bones;
#pragma unroll
    for (int e = 0; e < 16; ++e) bones[e] = (_Float16)1.0f;

    v8f acc[5] = {};   // acc[0..3]: 4 N-tiles of output; acc[4]: row counts

    for (int kb = 0; kb < K_DIM; kb += 32) {
        __builtin_prefetch(pRow + kb + 64, 0, 0);   // global_prefetch_b8; speculative, fault-silent

        // ---- A fragment: 16x32 f16 0/1 mask, contiguous k order (k = 16g + s) ----
        // lane group g holds K run [16g, 16g+16)
        const float4* p0 = (const float4*)(pRow + kb + 16 * g);
        const float4* p1 = (const float4*)(pRow + kb + 16 * g + 8);
        const float4  f0 = p0[0], f1 = p0[1];
        const float4  f2 = p1[0], f3 = p1[1];
        v16h a;
        a[0]  = (_Float16)(f0.x > 0.5f);  a[1]  = (_Float16)(f0.y > 0.5f);
        a[2]  = (_Float16)(f0.z > 0.5f);  a[3]  = (_Float16)(f0.w > 0.5f);
        a[4]  = (_Float16)(f1.x > 0.5f);  a[5]  = (_Float16)(f1.y > 0.5f);
        a[6]  = (_Float16)(f1.z > 0.5f);  a[7]  = (_Float16)(f1.w > 0.5f);
        a[8]  = (_Float16)(f2.x > 0.5f);  a[9]  = (_Float16)(f2.y > 0.5f);
        a[10] = (_Float16)(f2.z > 0.5f);  a[11] = (_Float16)(f2.w > 0.5f);
        a[12] = (_Float16)(f3.x > 0.5f);  a[13] = (_Float16)(f3.y > 0.5f);
        a[14] = (_Float16)(f3.z > 0.5f);  a[15] = (_Float16)(f3.w > 0.5f);

        // ---- 4 N-tiles: B fragment is 16 contiguous f16 in LDS (32B aligned) ----
        // b[e] = emb[kb + 16g + e][nt*16 + n]  == same contiguous k order as A (lane=N, lanes 16-31 = K+16)
#pragma unroll
        for (int nt = 0; nt < 4; ++nt) {
            const v16h b = *(const v16h*)&ldsEmbT[(nt * 16 + n) * K_DIM + kb + 16 * g];
            acc[nt] = __builtin_amdgcn_wmma_f32_16x16x32_f16(
                          false, a, false, b, (short)0, acc[nt], false, false);
        }
        // ---- Count WMMA: mask @ ones ----
        acc[4] = __builtin_amdgcn_wmma_f32_16x16x32_f16(
                     false, a, false, bones, (short)0, acc[4], false, false);
    }

    // C/D layout: VGPR r of lane -> row (8g + r), col n. acc[4][r] = count of that same row.
#pragma unroll
    for (int r = 0; r < 8; ++r) {
        const float c   = acc[4][r];                       // integer-valued f32, exact
        const float inv = (c > 0.0f) ? __builtin_amdgcn_rcpf(c) : 0.0f;
        float* po = out + (size_t)(row0 + 8 * g + r) * D_DIM + n;
#pragma unroll
        for (int nt = 0; nt < 4; ++nt) {
            po[nt * 16] = acc[nt][r] * inv;   // 16 lanes -> 64B coalesced per row-segment
        }
    }
}

extern "C" void kernel_launch(void* const* d_in, const int* in_sizes, int n_in,
                              void* d_out, int out_size, void* d_ws, size_t ws_size,
                              hipStream_t stream) {
    const float* flags = (const float*)d_in[0];   // [T, 512] f32
    const float* emb   = (const float*)d_in[1];   // [512, 64] f32
    float*       out   = (float*)d_out;           // [T, 64] f32

    const int T        = in_sizes[0] / K_DIM;     // 100000
    const int numTiles = T >> 4;                  // 6250
    const int blocks   = (numTiles + WAVES_PER_BLOCK - 1) / WAVES_PER_BLOCK;

    FlagBagEncoder_wmma_kernel<<<blocks, BLOCK_THREADS, 0, stream>>>(flags, emb, out, T);
}
